// GNN_Survival_47863115546736
// MI455X (gfx1250) — matrix operand order repaired and checked by reading, not compiled, not run on hardware
//
#include <hip/hip_runtime.h>
#include <hip/hip_bf16.h>
#include <stdint.h>
#include <stddef.h>

#define HEADS 8
#define HID   128
#define NA    10000
#define NB    10000
#define NE    120000
#define NGR   8
#define FA    256
#define FB    128

typedef __attribute__((ext_vector_type(16))) __bf16 v16bf;
typedef __attribute__((ext_vector_type(8)))  __bf16 v8bf;
typedef __attribute__((ext_vector_type(8)))  float  v8f;

__device__ __forceinline__ unsigned short f2bf(float f) {
    union { float f; unsigned u; } v; v.f = f;
    unsigned u = v.u + 0x7FFFu + ((v.u >> 16) & 1u);   // round-to-nearest-even
    return (unsigned short)(u >> 16);
}

// ---------------------------------------------------------------- utilities
__global__ void fill_zero_kernel(unsigned* p, size_t n) {
    size_t i = (size_t)blockIdx.x * blockDim.x + threadIdx.x;
    size_t st = (size_t)gridDim.x * blockDim.x;
    for (; i < n; i += st) p[i] = 0u;
}

__global__ void f32_to_bf16_kernel(const float* in, unsigned short* out, size_t n) {
    size_t i = (size_t)blockIdx.x * blockDim.x + threadIdx.x;
    size_t st = (size_t)gridDim.x * blockDim.x;
    for (; i < n; i += st) out[i] = f2bf(in[i]);
}

// Pack row-major f32 weight [K,N] into per-wave B fragments (bf16).
// Fragment block (nt,kt): 32 lanes x 16 contiguous bf16.
// lane = kgroup*16 + (n%16); element e -> K = kt*32 + kgroup*8 + (e&7) + (e>>3)*16.
__global__ void pack_weight_kernel(const float* W, unsigned short* P, int K, int N) {
    int Ktiles = K >> 5;
    size_t total = (size_t)K * N;
    size_t i = (size_t)blockIdx.x * blockDim.x + threadIdx.x;
    size_t st = (size_t)gridDim.x * blockDim.x;
    for (; i < total; i += st) {
        int e    = (int)(i & 15);
        int lane = (int)((i >> 4) & 31);
        int blk  = (int)(i >> 9);
        int kt   = blk % Ktiles;
        int nt   = blk / Ktiles;
        int n = nt * 16 + (lane & 15);
        int k = kt * 32 + ((lane >> 4) << 3) + (e & 7) + ((e >> 3) << 4);
        P[i] = f2bf(W[(size_t)k * N + n]);
    }
}

// ------------------------------------------------------------- WMMA GEMM
// C[M,N] = A[M,K](bf16, row-major) @ Bpacked + bias, optional relu,
// optional bf16 mirror. One wave = 16(M) x (16*NTW)(N) tile.
template<int NTW>
__global__ void gemm_bf16_kernel(const unsigned short* __restrict__ A,
                                 const unsigned short* __restrict__ Bp,
                                 const float* __restrict__ bias,
                                 float* __restrict__ C,
                                 unsigned short* __restrict__ Cbf,
                                 int M, int N, int K, int relu) {
    const int Ktiles   = K >> 5;
    const int NtGroups = (N >> 4) / NTW;
    const int wave = blockIdx.x * (blockDim.x >> 5) + (threadIdx.x >> 5);
    const int totalWaves = (M >> 4) * NtGroups;
    if (wave >= totalWaves) return;                 // wave-uniform: EXEC stays all-ones
    const int mt = wave / NtGroups;
    const int ng = wave % NtGroups;
    const int lane = threadIdx.x & 31;
    const int half = lane >> 4;                     // 0: K[0..7],[16..23] ; 1: +8
    const int lr   = lane & 15;

    const unsigned short* Arow = A + (size_t)(mt * 16 + lr) * K + half * 8;

    v8f acc[NTW];
#pragma unroll
    for (int i = 0; i < NTW; ++i) acc[i] = (v8f){0.f,0.f,0.f,0.f,0.f,0.f,0.f,0.f};

    for (int kt = 0; kt < Ktiles; ++kt) {
        v8bf lo = *reinterpret_cast<const v8bf*>(Arow + kt * 32);
        v8bf hi = *reinterpret_cast<const v8bf*>(Arow + kt * 32 + 16);
        v16bf a = __builtin_shufflevector(lo, hi, 0,1,2,3,4,5,6,7,8,9,10,11,12,13,14,15);
#pragma unroll
        for (int i = 0; i < NTW; ++i) {
            int nt = ng * NTW + i;
            v16bf b = *reinterpret_cast<const v16bf*>(
                Bp + ((size_t)(nt * Ktiles + kt) * 32 + lane) * 16);
            acc[i] = __builtin_amdgcn_wmma_f32_16x16x32_bf16(
                false, a, false, b, (short)0, acc[i], false, false);
        }
    }
#pragma unroll
    for (int i = 0; i < NTW; ++i) {
        int nt  = ng * NTW + i;
        int col = nt * 16 + lr;
        float bv = bias ? bias[col] : 0.f;
#pragma unroll
        for (int v = 0; v < 8; ++v) {
            int row = mt * 16 + v + half * 8;       // C layout: lanes 16-31 -> M+8
            float val = acc[i][v] + bv;
            if (relu) val = val > 0.f ? val : 0.f;
            C[(size_t)row * N + col] = val;
            if (Cbf) Cbf[(size_t)row * N + col] = f2bf(val);
        }
    }
}

// ----------------------------------------------------- attention pieces
// al[n,h] = sum_c h[n,h,c] * avec[h,c].  block=256 (8 waves = 8 heads), grid=N
__global__ void attn_logits_kernel(const float* __restrict__ h,
                                   const float* __restrict__ avec,
                                   float* __restrict__ al, int Nn) {
    int node = blockIdx.x;
    int head = threadIdx.x >> 5;
    int lane = threadIdx.x & 31;
    const float4 hv = *reinterpret_cast<const float4*>(
        h + (size_t)node * (HEADS * HID) + head * HID + lane * 4);
    const float4 av = *reinterpret_cast<const float4*>(avec + head * HID + lane * 4);
    float s = hv.x * av.x + hv.y * av.y + hv.z * av.z + hv.w * av.w;
#pragma unroll
    for (int off = 16; off; off >>= 1) s += __shfl_xor(s, off, 32);
    if (lane == 0) al[(size_t)node * HEADS + head] = s;
}

__device__ __forceinline__ unsigned enc_f32(float f) {
    union { float f; unsigned u; } c; c.f = f;
    return (c.u & 0x80000000u) ? ~c.u : (c.u | 0x80000000u);
}
__device__ __forceinline__ float dec_f32(unsigned u) {
    union { unsigned u; float f; } c;
    c.u = (u & 0x80000000u) ? (u & 0x7FFFFFFFu) : ~u;
    return c.f;
}

// pass 1: e = leaky_relu(al_s[src]+al_d[dst]); segment max via uint atomicMax
__global__ void edge_logits_kernel(const int* __restrict__ src, const int* __restrict__ dst,
                                   const float* __restrict__ als, const float* __restrict__ ald,
                                   float* __restrict__ ebuf, unsigned* __restrict__ mmax) {
    size_t i = (size_t)blockIdx.x * blockDim.x + threadIdx.x;
    size_t st = (size_t)gridDim.x * blockDim.x;
    for (; i < (size_t)NE * HEADS; i += st) {
        int e = (int)(i >> 3), h = (int)(i & 7);
        int s = src[e], d = dst[e];
        float v = als[(size_t)s * HEADS + h] + ald[(size_t)d * HEADS + h];
        v = v > 0.f ? v : 0.2f * v;
        ebuf[i] = v;
        atomicMax(&mmax[(size_t)d * HEADS + h], enc_f32(v));
    }
}

// pass 2: ex = exp(e - m[dst]); segment sum
__global__ void edge_exp_kernel(const int* __restrict__ dst, float* __restrict__ ebuf,
                                const unsigned* __restrict__ mmax, float* __restrict__ denom) {
    size_t i = (size_t)blockIdx.x * blockDim.x + threadIdx.x;
    size_t st = (size_t)gridDim.x * blockDim.x;
    for (; i < (size_t)NE * HEADS; i += st) {
        int e = (int)(i >> 3), h = (int)(i & 7);
        int d = dst[e];
        float ex = __expf(ebuf[i] - dec_f32(mmax[(size_t)d * HEADS + h]));
        ebuf[i] = ex;
        atomicAdd(&denom[(size_t)d * HEADS + h], ex);
    }
}

// pass 3: acc[dst,h,c] += hs[src,h,c] * alpha.  block=256 per edge, wave=head.
__global__ void edge_scatter_kernel(const int* __restrict__ src, const int* __restrict__ dst,
                                    const float* __restrict__ ebuf, const float* __restrict__ denom,
                                    const float* __restrict__ hs, float* __restrict__ acc) {
    int e = blockIdx.x;
    int head = threadIdx.x >> 5;
    int lane = threadIdx.x & 31;
    int s = src[e], d = dst[e];
    float alpha = ebuf[(size_t)e * HEADS + head] /
                  (denom[(size_t)d * HEADS + head] + 1e-16f);
    const float4 hv = *reinterpret_cast<const float4*>(
        hs + (size_t)s * (HEADS * HID) + head * HID + lane * 4);
    float* o = acc + (size_t)d * (HEADS * HID) + head * HID + lane * 4;
    atomicAdd(o + 0, hv.x * alpha);
    atomicAdd(o + 1, hv.y * alpha);
    atomicAdd(o + 2, hv.z * alpha);
    atomicAdd(o + 3, hv.w * alpha);
}

// head-mean + bias (+relu), optional bf16 mirror
__global__ void gat_finalize_kernel(const float* __restrict__ acc, const float* __restrict__ bias,
                                    float* __restrict__ out, unsigned short* __restrict__ outbf,
                                    int Nn, int relu) {
    size_t i = (size_t)blockIdx.x * blockDim.x + threadIdx.x;
    size_t st = (size_t)gridDim.x * blockDim.x;
    for (; i < (size_t)Nn * HID; i += st) {
        int n = (int)(i >> 7), c = (int)(i & 127);
        float s = 0.f;
#pragma unroll
        for (int h = 0; h < HEADS; ++h) s += acc[(size_t)n * (HEADS * HID) + h * HID + c];
        s = s * 0.125f + bias[c];
        if (relu) s = s > 0.f ? s : 0.f;
        out[i] = s;
        if (outbf) outbf[i] = f2bf(s);
    }
}

// -------------------------------------------------------------- pooling/FC
__global__ void pool_acc_kernel(const float* __restrict__ h, const int* __restrict__ batch,
                                float* __restrict__ pool, float* __restrict__ cnt, int Nn) {
    size_t i = (size_t)blockIdx.x * blockDim.x + threadIdx.x;
    size_t st = (size_t)gridDim.x * blockDim.x;
    for (; i < (size_t)Nn * HID; i += st) {
        int n = (int)(i >> 7), c = (int)(i & 127);
        int g = batch[n];
        atomicAdd(&pool[(size_t)g * HID + c], h[i]);
        if (c == 0) atomicAdd(&cnt[g], 1.f);
    }
}

__global__ void final_fc_kernel(const float* __restrict__ poolA, const float* __restrict__ poolB,
                                const float* __restrict__ cntA, const float* __restrict__ cntB,
                                const float* __restrict__ fcw, const float* __restrict__ fcb,
                                float* __restrict__ out) {
    int g = blockIdx.x;
    int c = threadIdx.x;                 // 256 threads = 2*HID
    float v;
    if (c < HID) v = poolA[(size_t)g * HID + c] / fmaxf(cntA[g], 1.f);
    else         v = poolB[(size_t)g * HID + (c - HID)] / fmaxf(cntB[g], 1.f);
    v *= fcw[c];
    __shared__ float red[256];
    red[c] = v; __syncthreads();
    for (int s = 128; s; s >>= 1) { if (c < s) red[c] += red[c + s]; __syncthreads(); }
    if (c == 0) out[g] = red[0] + fcb[0];
}

// ----------------------------------------------------------------- driver
static void zero_f(void* p, size_t n_u32, hipStream_t s) {
    int blocks = (int)((n_u32 + 255) / 256); if (blocks > 4096) blocks = 4096;
    fill_zero_kernel<<<blocks, 256, 0, s>>>((unsigned*)p, n_u32);
}

static void run_gat(const unsigned short* hsrc_bf, const unsigned short* hdst_bf,
                    const unsigned short* pws, const unsigned short* pwd,
                    const float* as_vec, const float* ad_vec, const float* bias,
                    const int* src, const int* dst, int Nsrc, int Ndst,
                    float* hs, float* hd_acc, float* als, float* ald,
                    unsigned* mmax, float* denom, float* ebuf,
                    float* out_f32, unsigned short* out_bf, int relu, hipStream_t s) {
    // hs = hsrc @ ws ; hd = hdst @ wd   (M=10000, N=1024, K=128)
    {
        int waves = (Nsrc >> 4) * ((1024 >> 4) / 4);
        gemm_bf16_kernel<4><<<(waves + 7) / 8, 256, 0, s>>>(hsrc_bf, pws, nullptr, hs, nullptr, Nsrc, 1024, 128, 0);
        waves = (Ndst >> 4) * ((1024 >> 4) / 4);
        gemm_bf16_kernel<4><<<(waves + 7) / 8, 256, 0, s>>>(hdst_bf, pwd, nullptr, hd_acc, nullptr, Ndst, 1024, 128, 0);
    }
    attn_logits_kernel<<<Nsrc, 256, 0, s>>>(hs, as_vec, als, Nsrc);
    attn_logits_kernel<<<Ndst, 256, 0, s>>>(hd_acc, ad_vec, ald, Ndst);
    zero_f(mmax,  (size_t)Ndst * HEADS, s);
    zero_f(denom, (size_t)Ndst * HEADS, s);
    edge_logits_kernel<<<3750, 256, 0, s>>>(src, dst, als, ald, ebuf, mmax);
    edge_exp_kernel<<<3750, 256, 0, s>>>(dst, ebuf, mmax, denom);
    zero_f(hd_acc, (size_t)Ndst * HEADS * HID, s);     // hd dead; reuse as accumulator
    edge_scatter_kernel<<<NE, 256, 0, s>>>(src, dst, ebuf, denom, hs, hd_acc);
    gat_finalize_kernel<<<2048, 256, 0, s>>>(hd_acc, bias, out_f32, out_bf, Ndst, relu);
}

extern "C" void kernel_launch(void* const* d_in, const int* in_sizes, int n_in,
                              void* d_out, int out_size, void* d_ws, size_t ws_size,
                              hipStream_t stream) {
    const float* x_a    = (const float*)d_in[0];
    const float* x_b    = (const float*)d_in[1];
    const int* edge_ab  = (const int*)d_in[2];
    const int* edge_ba  = (const int*)d_in[3];
    const int* batch_a  = (const int*)d_in[4];
    const int* batch_b  = (const int*)d_in[5];
    const float* lin_a_w = (const float*)d_in[6];
    const float* lin_a_b = (const float*)d_in[7];
    const float* lin_b_w = (const float*)d_in[8];
    const float* lin_b_b = (const float*)d_in[9];
    const float* fc_w    = (const float*)d_in[10];
    const float* fc_b    = (const float*)d_in[11];
    // 12..31: {w,wd,as,ad,b} for (1,ab),(1,ba),(2,ab),(2,ba)
    const float* Wg[8]; const float* Ag[8]; const float* Bg[4];
    for (int i = 0; i < 4; ++i) {
        Wg[2*i + 0] = (const float*)d_in[12 + 5*i + 0];   // ws
        Wg[2*i + 1] = (const float*)d_in[12 + 5*i + 1];   // wd
        Ag[2*i + 0] = (const float*)d_in[12 + 5*i + 2];   // a_s
        Ag[2*i + 1] = (const float*)d_in[12 + 5*i + 3];   // a_d
        Bg[i]       = (const float*)d_in[12 + 5*i + 4];   // bias
    }
    const int* src_ab = edge_ab;         const int* dst_ab = edge_ab + NE;
    const int* src_ba = edge_ba;         const int* dst_ba = edge_ba + NE;

    // ---- workspace carving (256B aligned)
    char* base = (char*)d_ws; size_t off = 0;
    auto carve = [&](size_t bytes) { char* p = base + off; off += (bytes + 255) & ~(size_t)255; return p; };
    unsigned short* xa_bf   = (unsigned short*)carve((size_t)NA * FA * 2);
    unsigned short* xb_bf   = (unsigned short*)carve((size_t)NB * FB * 2);
    unsigned short* pw_la   = (unsigned short*)carve((size_t)FA * HID * 2);
    unsigned short* pw_lb   = (unsigned short*)carve((size_t)FB * HID * 2);
    unsigned short* pw_g[8];
    for (int i = 0; i < 8; ++i) pw_g[i] = (unsigned short*)carve((size_t)HID * (HEADS * HID) * 2);
    float* ha  = (float*)carve((size_t)NA * HID * 4);
    float* hb  = (float*)carve((size_t)NB * HID * 4);
    unsigned short* ha_bf = (unsigned short*)carve((size_t)NA * HID * 2);
    unsigned short* hb_bf = (unsigned short*)carve((size_t)NB * HID * 2);
    float* h1a = (float*)carve((size_t)NA * HID * 4);
    float* h1b = (float*)carve((size_t)NB * HID * 4);
    unsigned short* h1a_bf = (unsigned short*)carve((size_t)NA * HID * 2);
    unsigned short* h1b_bf = (unsigned short*)carve((size_t)NB * HID * 2);
    float* h2a = (float*)carve((size_t)NA * HID * 4);
    float* h2b = (float*)carve((size_t)NB * HID * 4);
    float* hs     = (float*)carve((size_t)NA * HEADS * HID * 4);   // 41 MB
    float* hd_acc = (float*)carve((size_t)NA * HEADS * HID * 4);   // 41 MB (hd -> acc reuse)
    float* als = (float*)carve((size_t)NA * HEADS * 4);
    float* ald = (float*)carve((size_t)NA * HEADS * 4);
    unsigned* mmax = (unsigned*)carve((size_t)NA * HEADS * 4);
    float* denom   = (float*)carve((size_t)NA * HEADS * 4);
    float* ebuf    = (float*)carve((size_t)NE * HEADS * 4);
    float* poolA = (float*)carve((size_t)NGR * HID * 4);
    float* poolB = (float*)carve((size_t)NGR * HID * 4);
    float* cntA  = (float*)carve((size_t)NGR * 4);
    float* cntB  = (float*)carve((size_t)NGR * 4);
    (void)ws_size; (void)n_in; (void)in_sizes; (void)out_size;

    // ---- pack weights (bf16 fragment layout), convert inputs
    pack_weight_kernel<<<128, 256, 0, stream>>>(lin_a_w, pw_la, FA, HID);
    pack_weight_kernel<<<64, 256, 0, stream>>>(lin_b_w, pw_lb, FB, HID);
    for (int i = 0; i < 8; ++i)
        pack_weight_kernel<<<512, 256, 0, stream>>>(Wg[i], pw_g[i], HID, HEADS * HID);
    f32_to_bf16_kernel<<<4096, 256, 0, stream>>>(x_a, xa_bf, (size_t)NA * FA);
    f32_to_bf16_kernel<<<4096, 256, 0, stream>>>(x_b, xb_bf, (size_t)NB * FB);

    // ---- input linears: h = relu(x @ W + b)
    {
        int waves = (NA >> 4) * ((HID >> 4) / 4);
        gemm_bf16_kernel<4><<<(waves + 7) / 8, 256, 0, stream>>>(xa_bf, pw_la, lin_a_b, ha, ha_bf, NA, HID, FA, 1);
        waves = (NB >> 4) * ((HID >> 4) / 4);
        gemm_bf16_kernel<4><<<(waves + 7) / 8, 256, 0, stream>>>(xb_bf, pw_lb, lin_b_b, hb, hb_bf, NB, HID, FB, 1);
    }

    // ---- layer 1: hb1 = GAT_ab(ha->hb), ha1 = GAT_ba(hb->ha); both relu'd
    run_gat(ha_bf, hb_bf, pw_g[0], pw_g[1], Ag[0], Ag[1], Bg[0], src_ab, dst_ab, NA, NB,
            hs, hd_acc, als, ald, mmax, denom, ebuf, h1b, h1b_bf, 1, stream);
    run_gat(hb_bf, ha_bf, pw_g[2], pw_g[3], Ag[2], Ag[3], Bg[1], src_ba, dst_ba, NB, NA,
            hs, hd_acc, als, ald, mmax, denom, ebuf, h1a, h1a_bf, 1, stream);

    // ---- layer 2 (no relu)
    run_gat(h1a_bf, h1b_bf, pw_g[4], pw_g[5], Ag[4], Ag[5], Bg[2], src_ab, dst_ab, NA, NB,
            hs, hd_acc, als, ald, mmax, denom, ebuf, h2b, nullptr, 0, stream);
    run_gat(h1b_bf, h1a_bf, pw_g[6], pw_g[7], Ag[6], Ag[7], Bg[3], src_ba, dst_ba, NB, NA,
            hs, hd_acc, als, ald, mmax, denom, ebuf, h2a, nullptr, 0, stream);

    // ---- mean pool + FC
    zero_f(poolA, (size_t)NGR * HID, stream);
    zero_f(poolB, (size_t)NGR * HID, stream);
    zero_f(cntA, NGR, stream);
    zero_f(cntB, NGR, stream);
    pool_acc_kernel<<<2048, 256, 0, stream>>>(h2a, batch_a, poolA, cntA, NA);
    pool_acc_kernel<<<2048, 256, 0, stream>>>(h2b, batch_b, poolB, cntB, NB);
    final_fc_kernel<<<NGR, 256, 0, stream>>>(poolA, poolB, cntA, cntB, fc_w, fc_b, (float*)d_out);
}
